// HookedMambaBlock_57518202028337
// MI455X (gfx1250) — compile-verified, matrix operand order checked by reading, not run on hardware
//
#include <hip/hip_runtime.h>
#include <hip/hip_bf16.h>
#include <math.h>

// Model dims (fixed by the reference)
#define DM   1024   // d_model
#define EI   2048   // d_inner
#define NS   16     // d_state
#define DDR  64     // dt_rank
#define KC   4      // d_conv
#define BB   2
#define LL   1024
#define MM   (BB*LL)  // 2048 rows

typedef __attribute__((ext_vector_type(16))) __bf16 v16bf;
typedef __attribute__((ext_vector_type(8)))  __bf16 v8bf;
typedef __attribute__((ext_vector_type(8)))  float  v8f;
typedef unsigned short u16;   // bf16 storage

// ---------------- CDNA5 async-copy primitives ----------------
// global_load_async_to_lds_b128: per-lane 16B global->LDS DMA, tracked by ASYNCcnt.
__device__ __forceinline__ void async_cp16(unsigned int lds_off, const u16* g) {
  asm volatile("global_load_async_to_lds_b128 %0, %1, off"
               :: "v"(lds_off), "v"(g) : "memory");
}
#if __has_builtin(__builtin_amdgcn_s_wait_asynccnt)
#define WAIT_ASYNC(n) __builtin_amdgcn_s_wait_asynccnt(n)
#else
#define WAIT_ASYNC(n) asm volatile("s_wait_asynccnt " #n ::: "memory")
#endif

// ---------------- device helpers ----------------
__device__ __forceinline__ u16 f2b(float f) {
  __hip_bfloat16 h = __float2bfloat16(f);   // RNE convert
  u16 r; __builtin_memcpy(&r, &h, 2); return r;
}
__device__ __forceinline__ float silu(float x) {
  return x / (1.f + __expf(-x));
}
// Load a 32B WMMA fragment from LDS as two 16B-aligned halves (ds_load_b128 x2).
__device__ __forceinline__ v16bf lds_frag(const char* p) {
  v8bf lo = *(const v8bf*)(p);
  v8bf hi = *(const v8bf*)(p + 16);
  return __builtin_shufflevector(lo, hi, 0,1,2,3,4,5,6,7,8,9,10,11,12,13,14,15);
}

// ---------------- elementwise kernels ----------------
__global__ __launch_bounds__(256) void k_cast_bf16(const float* __restrict__ in,
                                                   u16* __restrict__ out, int n) {
  int i = blockIdx.x * 256 + threadIdx.x;
  if (i < n) out[i] = f2b(in[i]);
}

__global__ __launch_bounds__(256) void k_negexp(const float* __restrict__ a_log,
                                                float* __restrict__ a_neg, int n) {
  int i = blockIdx.x * 256 + threadIdx.x;
  if (i < n) a_neg[i] = -__expf(a_log[i]);
}

// RMSNorm over D, emit bf16 activations for the WMMA GEMMs. One block per row.
__global__ __launch_bounds__(256) void k_rmsnorm(const float* __restrict__ resid,
                                                 const float* __restrict__ norm_w,
                                                 u16* __restrict__ xbf) {
  __shared__ float red[256];
  const int row = blockIdx.x;
  const float* rp = resid + (size_t)row * DM;
  float s = 0.f;
  for (int i = threadIdx.x; i < DM; i += 256) { float v = rp[i]; s += v * v; }
  red[threadIdx.x] = s;
  __syncthreads();
  for (int off = 128; off > 0; off >>= 1) {
    if (threadIdx.x < off) red[threadIdx.x] += red[threadIdx.x + off];
    __syncthreads();
  }
  const float scale = rsqrtf(red[0] * (1.f / DM) + 1e-5f);
  for (int i = threadIdx.x; i < DM; i += 256)
    xbf[(size_t)row * DM + i] = f2b(rp[i] * scale * norm_w[i]);
}

// Depthwise causal conv1d (K=4) + bias + silu; emits f32 and bf16 copies of x2.
__global__ __launch_bounds__(256) void k_conv_silu(const float* __restrict__ xi,
                                                   const float* __restrict__ cw,
                                                   const float* __restrict__ cb,
                                                   float* __restrict__ x2,
                                                   u16* __restrict__ x2b) {
  int idx = blockIdx.x * 256 + threadIdx.x;   // over MM*EI, [m, e] row-major
  if (idx >= MM * EI) return;
  const int e = idx % EI;
  const int m = idx / EI;
  const int l = m % LL;
  float s = cb[e];
#pragma unroll
  for (int j = 0; j < KC; ++j) {
    int lj = l - (KC - 1) + j;
    if (lj >= 0)
      s += cw[e * KC + j] * xi[(size_t)(m - (KC - 1) + j) * EI + e];
  }
  float v = silu(s);
  x2[idx] = v;
  x2b[idx] = f2b(v);
}

// Selective scan: one thread per (b, e) channel, h[NS] in registers, loop over L.
__global__ __launch_bounds__(256) void k_scan(const float* __restrict__ delta,
                                              const float* __restrict__ x2,
                                              const float* __restrict__ Bm,
                                              const float* __restrict__ Cm,
                                              const float* __restrict__ A_neg,
                                              float* __restrict__ y) {
  int t = blockIdx.x * 256 + threadIdx.x;     // over BB*EI
  if (t >= BB * EI) return;
  const int e = t % EI;
  const int b = t / EI;
  float an[NS], h[NS];
#pragma unroll
  for (int n = 0; n < NS; ++n) { an[n] = A_neg[e * NS + n]; h[n] = 0.f; }
  const size_t base = (size_t)b * LL;
  for (int l = 0; l < LL; ++l) {
    const size_t m = base + l;
    const float d  = delta[m * EI + e];
    const float xv = x2[m * EI + e];
    const float* bp = Bm + m * NS;             // broadcast across the wave
    const float* cp = Cm + m * NS;
    float acc = 0.f;
#pragma unroll
    for (int n = 0; n < NS; ++n) {
      h[n] = __expf(d * an[n]) * h[n] + d * bp[n] * xv;
      acc += h[n] * cp[n];
    }
    y[m * EI + e] = acc;
  }
}

// y_gated(bf16) = (y + x2*W_D[e]) * silu(skip)
__global__ __launch_bounds__(256) void k_gate(const float* __restrict__ y,
                                              const float* __restrict__ x2,
                                              const float* __restrict__ W_D,
                                              const float* __restrict__ skip,
                                              u16* __restrict__ ygb) {
  int idx = blockIdx.x * 256 + threadIdx.x;
  if (idx >= MM * EI) return;
  const int e = idx % EI;
  ygb[idx] = f2b((y[idx] + x2[idx] * W_D[e]) * silu(skip[idx]));
}

// ---------------- shared epilogue store ----------------
//   0: plain store   1: +bias[col], softplus   2: += resid[row,col]
__device__ __forceinline__ void store_tile(v8f acc, float* C, int ldc,
                                           int M, int N, int row0, int col0,
                                           int lane, int ep,
                                           const float* bias,
                                           const float* resid) {
  const int col = col0 + (lane & 15);
  if (col >= N) return;
  const int rbase = row0 + ((lane >> 4) * 8);
#pragma unroll
  for (int r = 0; r < 8; ++r) {
    const int row = rbase + r;
    if (row >= M) continue;
    float v = acc[r];
    const size_t idx = (size_t)row * ldc + col;
    if (ep == 1)      { v += bias[col]; v = (v > 20.f) ? v : log1pf(__expf(v)); }
    else if (ep == 2) { v += resid[idx]; }
    C[idx] = v;
  }
}

// ---------------- direct-load WMMA GEMM (for skinny N: 16/64) ----------------
__global__ __launch_bounds__(256) void k_gemm_bf16(const u16* __restrict__ A, int lda,
                                                   const u16* __restrict__ W, int ldw,
                                                   float* __restrict__ C, int ldc,
                                                   int M, int N, int Ksz,
                                                   int epilogue,
                                                   const float* __restrict__ bias,
                                                   const float* __restrict__ resid) {
  const int lane = threadIdx.x & 31;
  const int wave = threadIdx.x >> 5;
  const int tiles_n = (N + 31) >> 5;
  const int tiles_m = (M + 31) >> 5;
  const int gw = blockIdx.x * 8 + wave;
  if (gw >= tiles_m * tiles_n) return;        // wave-uniform exit
  const int row0 = (gw / tiles_n) * 32;
  const int col0 = (gw % tiles_n) * 32;

  const int fm = lane & 15;                   // fragment row/col within 16
  const int kh = lane >> 4;                   // which K-half this lane holds

  int ra0 = row0 + fm;      if (ra0 > M - 1) ra0 = M - 1;
  int ra1 = row0 + 16 + fm; if (ra1 > M - 1) ra1 = M - 1;
  int cb0 = col0 + fm;      if (cb0 > N - 1) cb0 = N - 1;
  int cb1 = col0 + 16 + fm; if (cb1 > N - 1) cb1 = N - 1;

  const u16* pa0 = A + (size_t)ra0 * lda + kh * 16;
  const u16* pa1 = A + (size_t)ra1 * lda + kh * 16;
  const u16* pb0 = W + (size_t)cb0 * ldw + kh * 16;
  const u16* pb1 = W + (size_t)cb1 * ldw + kh * 16;

  v8f acc00 = {}, acc01 = {}, acc10 = {}, acc11 = {};
  for (int k = 0; k < Ksz; k += 32) {
    v16bf a0 = *(const v16bf*)(pa0 + k);
    v16bf a1 = *(const v16bf*)(pa1 + k);
    v16bf b0 = *(const v16bf*)(pb0 + k);
    v16bf b1 = *(const v16bf*)(pb1 + k);
    acc00 = __builtin_amdgcn_wmma_f32_16x16x32_bf16(false, a0, false, b0, (short)0, acc00, false, false);
    acc01 = __builtin_amdgcn_wmma_f32_16x16x32_bf16(false, a0, false, b1, (short)0, acc01, false, false);
    acc10 = __builtin_amdgcn_wmma_f32_16x16x32_bf16(false, a1, false, b0, (short)0, acc10, false, false);
    acc11 = __builtin_amdgcn_wmma_f32_16x16x32_bf16(false, a1, false, b1, (short)0, acc11, false, false);
  }
  store_tile(acc00, C, ldc, M, N, row0,      col0,      lane, epilogue, bias, resid);
  store_tile(acc01, C, ldc, M, N, row0,      col0 + 16, lane, epilogue, bias, resid);
  store_tile(acc10, C, ldc, M, N, row0 + 16, col0,      lane, epilogue, bias, resid);
  store_tile(acc11, C, ldc, M, N, row0 + 16, col0 + 16, lane, epilogue, bias, resid);
}

// ------------- async-LDS double-buffered WMMA GEMM (big GEMMs) -------------
// Block tile 128x128, 8 waves (2 M x 4 N), each wave 64x32 = 4x2 WMMA tiles.
// K staged in 32-deep slabs via global_load_async_to_lds_b128, double buffered.
#define GBM   128
#define GBN   128
#define LDSTR 80                  // padded LDS row stride: 16B aligned, conflict-free frags
#define ASLAB (GBM * LDSTR)       // 10240 B per slab
#define BUFSZ (2 * ASLAB)         // A + B slabs per buffer

__device__ __forceinline__ void stage_slabs(unsigned int smem_base, int buf,
                                            const u16* __restrict__ A, int lda,
                                            const u16* __restrict__ W, int ldw,
                                            int M, int N, int row0, int col0, int k,
                                            int wave, int lane) {
  const unsigned int base = smem_base + (unsigned int)buf * BUFSZ;
#pragma unroll
  for (int i = 0; i < 2; ++i) {
    const int q = wave * 64 + i * 32 + lane;   // 16B chunk id: 128 rows x 4 chunks
    const int r = q >> 2, c = q & 3;
    int gr = row0 + r; if (gr > M - 1) gr = M - 1;
    async_cp16(base + (unsigned int)(r * LDSTR + c * 16),
               A + (size_t)gr * lda + k + c * 8);
    int gc = col0 + r; if (gc > N - 1) gc = N - 1;
    async_cp16(base + ASLAB + (unsigned int)(r * LDSTR + c * 16),
               W + (size_t)gc * ldw + k + c * 8);
  }
}

__global__ __launch_bounds__(256) void k_gemm_bf16_lds(const u16* __restrict__ A, int lda,
                                                       const u16* __restrict__ W, int ldw,
                                                       float* __restrict__ C, int ldc,
                                                       int M, int N, int Ksz,
                                                       int epilogue,
                                                       const float* __restrict__ bias,
                                                       const float* __restrict__ resid) {
  __shared__ char smem[2 * BUFSZ];             // 40 KB of the 320 KB WGP LDS
  const int lane = threadIdx.x & 31;
  const int wave = threadIdx.x >> 5;
  const int wm = wave >> 2;                    // 0..1 : M offset 64*wm
  const int wn = wave & 3;                     // 0..3 : N offset 32*wn
  const int row0 = blockIdx.y * GBM;
  const int col0 = blockIdx.x * GBN;
  const int fm = lane & 15;
  const int kh = lane >> 4;
  const unsigned int smem_base = (unsigned int)(uintptr_t)smem;  // low 32b = LDS offset

  v8f zero = {};
  v8f acc[4][2];
#pragma unroll
  for (int i = 0; i < 4; ++i)
#pragma unroll
    for (int j = 0; j < 2; ++j) acc[i][j] = zero;

  const int nk = Ksz >> 5;
  stage_slabs(smem_base, 0, A, lda, W, ldw, M, N, row0, col0, 0, wave, lane);

  for (int ks = 0; ks < nk; ++ks) {
    if (ks + 1 < nk) {
      // overlap: issue next slab into the other buffer (4 async ops/wave)...
      stage_slabs(smem_base, (ks + 1) & 1, A, lda, W, ldw,
                  M, N, row0, col0, (ks + 1) << 5, wave, lane);
      WAIT_ASYNC(4);   // ...and wait only for the previous slab (in-order completion)
    } else {
      WAIT_ASYNC(0);
    }
    __syncthreads();   // all waves' staged data visible

    const char* ba = smem + (ks & 1) * BUFSZ;
    const char* bb = ba + ASLAB;
    v16bf a[4], b[2];
#pragma unroll
    for (int i = 0; i < 4; ++i)
      a[i] = lds_frag(ba + (wm * 64 + i * 16 + fm) * LDSTR + kh * 32);
#pragma unroll
    for (int j = 0; j < 2; ++j)
      b[j] = lds_frag(bb + (wn * 32 + j * 16 + fm) * LDSTR + kh * 32);
#pragma unroll
    for (int i = 0; i < 4; ++i)
#pragma unroll
      for (int j = 0; j < 2; ++j)
        acc[i][j] = __builtin_amdgcn_wmma_f32_16x16x32_bf16(
            false, a[i], false, b[j], (short)0, acc[i][j], false, false);
    __syncthreads();   // done reading this buffer before it is restaged
  }

#pragma unroll
  for (int i = 0; i < 4; ++i)
#pragma unroll
    for (int j = 0; j < 2; ++j)
      store_tile(acc[i][j], C, ldc, M, N, row0 + wm * 64 + i * 16,
                 col0 + wn * 32 + j * 16, lane, epilogue, bias, resid);
}

// ---------------- host side ----------------
static inline int cdiv(int a, int b) { return (a + b - 1) / b; }
static inline int gemm_grid(int M, int N) { return cdiv(cdiv(M, 32) * cdiv(N, 32), 8); }
static inline dim3 lds_grid(int M, int N) { return dim3((unsigned)cdiv(N, GBN), (unsigned)cdiv(M, GBM), 1); }

extern "C" void kernel_launch(void* const* d_in, const int* in_sizes, int n_in,
                              void* d_out, int out_size, void* d_ws, size_t ws_size,
                              hipStream_t stream) {
  (void)in_sizes; (void)n_in; (void)out_size; (void)ws_size;
  const float* resid  = (const float*)d_in[0];
  const float* norm_w = (const float*)d_in[2];
  const float* skip_w = (const float*)d_in[3];
  const float* in_w   = (const float*)d_in[4];
  const float* conv_w = (const float*)d_in[5];
  const float* conv_b = (const float*)d_in[6];
  const float* wd1    = (const float*)d_in[7];
  const float* wd2_w  = (const float*)d_in[8];
  const float* wd2_b  = (const float*)d_in[9];
  const float* wb     = (const float*)d_in[10];
  const float* wc     = (const float*)d_in[11];
  const float* A_log  = (const float*)d_in[12];
  const float* W_D    = (const float*)d_in[13];
  const float* out_w  = (const float*)d_in[14];
  float* out = (float*)d_out;

  // Carve scratch (~110 MB; fits comfortably in the MI455X 192 MB L2 once hot).
  char* p = (char*)d_ws;
  auto take = [&](size_t bytes) { char* r = p; p += (bytes + 255) & ~(size_t)255; return r; };
  u16*   x_bf    = (u16*)  take((size_t)MM * DM * 2);
  u16*   skipw_b = (u16*)  take((size_t)EI * DM * 2);
  u16*   inw_b   = (u16*)  take((size_t)EI * DM * 2);
  u16*   wd1_b   = (u16*)  take((size_t)DDR * EI * 2);
  u16*   wd2w_b  = (u16*)  take((size_t)EI * DDR * 2);
  u16*   wb_b    = (u16*)  take((size_t)NS * EI * 2);
  u16*   wc_b    = (u16*)  take((size_t)NS * EI * 2);
  u16*   outw_b  = (u16*)  take((size_t)DM * EI * 2);
  float* skip    = (float*)take((size_t)MM * EI * 4);
  float* xi      = (float*)take((size_t)MM * EI * 4);
  float* x2      = (float*)take((size_t)MM * EI * 4);
  u16*   x2_b    = (u16*)  take((size_t)MM * EI * 2);
  float* d1      = (float*)take((size_t)MM * DDR * 4);
  u16*   d1_b    = (u16*)  take((size_t)MM * DDR * 2);
  float* delta   = (float*)take((size_t)MM * EI * 4);
  float* Bmat    = (float*)take((size_t)MM * NS * 4);
  float* Cmat    = (float*)take((size_t)MM * NS * 4);
  float* A_neg   = (float*)take((size_t)EI * NS * 4);
  float* yacc    = (float*)take((size_t)MM * EI * 4);
  u16*   yg_b    = (u16*)  take((size_t)MM * EI * 2);

  // Weight down-converts (cheap, L2-resident afterwards)
  k_cast_bf16<<<cdiv(EI * DM, 256), 256, 0, stream>>>(skip_w, skipw_b, EI * DM);
  k_cast_bf16<<<cdiv(EI * DM, 256), 256, 0, stream>>>(in_w,   inw_b,   EI * DM);
  k_cast_bf16<<<cdiv(DDR * EI, 256), 256, 0, stream>>>(wd1,   wd1_b,   DDR * EI);
  k_cast_bf16<<<cdiv(EI * DDR, 256), 256, 0, stream>>>(wd2_w, wd2w_b,  EI * DDR);
  k_cast_bf16<<<cdiv(NS * EI, 256), 256, 0, stream>>>(wb,     wb_b,    NS * EI);
  k_cast_bf16<<<cdiv(NS * EI, 256), 256, 0, stream>>>(wc,     wc_b,    NS * EI);
  k_cast_bf16<<<cdiv(DM * EI, 256), 256, 0, stream>>>(out_w,  outw_b,  DM * EI);
  k_negexp   <<<cdiv(EI * NS, 256), 256, 0, stream>>>(A_log,  A_neg,   EI * NS);

  // 1) RMSNorm -> x (bf16)
  k_rmsnorm<<<MM, 256, 0, stream>>>(resid, norm_w, x_bf);

  // 2) skip = x @ skip_w^T ; xi = x @ in_w^T   (async-LDS WMMA)
  k_gemm_bf16_lds<<<lds_grid(MM, EI), 256, 0, stream>>>(x_bf, DM, skipw_b, DM, skip, EI, MM, EI, DM, 0, nullptr, nullptr);
  k_gemm_bf16_lds<<<lds_grid(MM, EI), 256, 0, stream>>>(x_bf, DM, inw_b,   DM, xi,   EI, MM, EI, DM, 0, nullptr, nullptr);

  // 3) depthwise causal conv + silu -> x2 (f32 + bf16)
  k_conv_silu<<<cdiv(MM * EI, 256), 256, 0, stream>>>(xi, conv_w, conv_b, x2, x2_b);

  // 4) delta = softplus(x2 @ wd1^T @ wd2^T + wd2_b)   (softplus fused in epilogue)
  k_gemm_bf16<<<gemm_grid(MM, DDR), 256, 0, stream>>>(x2_b, EI, wd1_b, EI, d1, DDR, MM, DDR, EI, 0, nullptr, nullptr);
  k_cast_bf16<<<cdiv(MM * DDR, 256), 256, 0, stream>>>(d1, d1_b, MM * DDR);
  k_gemm_bf16_lds<<<lds_grid(MM, EI), 256, 0, stream>>>(d1_b, DDR, wd2w_b, DDR, delta, EI, MM, EI, DDR, 1, wd2_b, nullptr);

  // 5) Bm, Cm = x2 @ wb^T / wc^T   (skinny N=16, direct-load WMMA)
  k_gemm_bf16<<<gemm_grid(MM, NS), 256, 0, stream>>>(x2_b, EI, wb_b, EI, Bmat, NS, MM, NS, EI, 0, nullptr, nullptr);
  k_gemm_bf16<<<gemm_grid(MM, NS), 256, 0, stream>>>(x2_b, EI, wc_b, EI, Cmat, NS, MM, NS, EI, 0, nullptr, nullptr);

  // 6) selective scan (sequential over L, parallel over B*E channels)
  k_scan<<<cdiv(BB * EI, 256), 256, 0, stream>>>(delta, x2, Bmat, Cmat, A_neg, yacc);

  // 7) gate: (y + x2*W_D) * silu(skip) -> bf16
  k_gate<<<cdiv(MM * EI, 256), 256, 0, stream>>>(yacc, x2, W_D, skip, yg_b);

  // 8) out = resid + y_g @ out_w^T   (residual fused in epilogue)
  k_gemm_bf16_lds<<<lds_grid(MM, DM), 256, 0, stream>>>(yg_b, EI, outw_b, EI, out, DM, MM, DM, EI, 2, nullptr, resid);
}
